// SparseMultiheadAttention_16303695856197
// MI455X (gfx1250) — matrix operand order repaired and checked by reading, not compiled, not run on hardware
//
#include <hip/hip_runtime.h>
#include <cstdint>
#include <cstddef>

// ---------------------------------------------------------------------------
// Sliding-window multi-head attention for MI455X (gfx1250, wave32, WMMA).
// Pipeline: f32->bf16 convert, WMMA bf16 GEMMs (QKV proj), WMMA windowed
// attention with f32 softmax, WMMA output GEMM (f32 out).
// ---------------------------------------------------------------------------

#define SEQL   2048
#define EMBED  768
#define HEADS  12
#define HD     64
#define SPAN   64
#define NWTILE 9      // ceil(144/16) window column tiles per 16-row block
#define LDSTR  160    // padded window width (5 K-chunks of 32)

typedef __attribute__((ext_vector_type(16))) __bf16 v16bf;
typedef __attribute__((ext_vector_type(8)))  __bf16 v8bf;
typedef __attribute__((ext_vector_type(8)))  float  v8f;

union BF16x16 { v16bf v; unsigned short s[16]; };

__device__ __forceinline__ unsigned short f2bf(float f) {
    union { float f; unsigned u; } x; x.f = f;
    unsigned r = x.u + 0x7FFFu + ((x.u >> 16) & 1u);   // round-to-nearest-even
    return (unsigned short)(r >> 16);
}

__device__ __forceinline__ v8f zero8() {
    v8f z;
#pragma unroll
    for (int i = 0; i < 8; ++i) z[i] = 0.0f;
    return z;
}

// A-fragment loader: lane holds row `arow`; contiguous 16-elem K run at 16*hi
__device__ __forceinline__ v16bf load_afrag(const unsigned short* __restrict__ A,
                                            int arow, int kc, int hi) {
    v8bf lo = *reinterpret_cast<const v8bf*>(A + arow + kc + 16 * hi);
    v8bf hh = *reinterpret_cast<const v8bf*>(A + arow + kc + 16 * hi + 8);
    return __builtin_shufflevector(lo, hh,
                0,1,2,3,4,5,6,7,8,9,10,11,12,13,14,15);
}

// ---------------------------------------------------------------------------
// f32 -> bf16 elementwise convert
// ---------------------------------------------------------------------------
__global__ void cvt_f32_bf16(const float* __restrict__ src,
                             unsigned short* __restrict__ dst, int n) {
    int i = blockIdx.x * blockDim.x + threadIdx.x;
    if (i < n) dst[i] = f2bf(src[i]);
}

// W [K][N] row-major f32 -> WT [N][K] bf16 (so B-fragment loads are contiguous)
__global__ void transpose_f32_bf16(const float* __restrict__ src,
                                   unsigned short* __restrict__ dst,
                                   int Kd, int Nd) {
    int i = blockIdx.x * blockDim.x + threadIdx.x;
    if (i < Kd * Nd) {
        int n = i / Kd, k = i % Kd;
        dst[i] = f2bf(src[k * Nd + n]);
    }
}

// ---------------------------------------------------------------------------
// GEMM: Out[M=2048][N=768] = A[2048][768](bf16) * W(via WT bf16) + bias(f32)
// One wave computes a 32x64 tile (2 M-tiles x 4 N-tiles): each B fragment is
// reused by 2 WMMAs -> ~1.5 b128 loads per v_wmma. 24 K-chunks * 8 = 192
// v_wmma per wave.
// mode 0: store bf16 row-major                       (Q, K projections)
// mode 1: store bf16 transposed [col][row], ld=SEQL  (V projection)
// mode 2: store f32  row-major                       (final output)
// ---------------------------------------------------------------------------
__global__ __launch_bounds__(128)
void gemm_wmma_bf16(const unsigned short* __restrict__ A,
                    const unsigned short* __restrict__ BT,
                    const float* __restrict__ bias,
                    void* __restrict__ Out, int mode) {
    const int lane = threadIdx.x & 31;
    const int wave = threadIdx.x >> 5;
    const int l15  = lane & 15;
    const int hi   = (lane >> 4) & 1;

    const int m0 = (blockIdx.x * 4 + wave) * 32;   // two 16-row tiles
    const int n0 = blockIdx.y * 64;

    v8f acc[2][4];
#pragma unroll
    for (int u = 0; u < 2; ++u)
#pragma unroll
        for (int t = 0; t < 4; ++t) acc[u][t] = zero8();

    const int arow0 = (m0 + l15) * EMBED;
    const int arow1 = arow0 + 16 * EMBED;
    const unsigned short* brow = BT + (size_t)(n0 + lane * 2) * EMBED;

    for (int kc = 0; kc < EMBED; kc += 32) {
        if (kc + 64 <= EMBED) {                    // one chunk ahead
            __builtin_prefetch(A + arow0 + kc + 32, 0, 3);
            __builtin_prefetch(A + arow1 + kc + 32, 0, 3);
            __builtin_prefetch(brow + kc + 32, 0, 3);
        }
        v16bf af0 = load_afrag(A, arow0, kc, hi);
        v16bf af1 = load_afrag(A, arow1, kc, hi);
#pragma unroll
        for (int t = 0; t < 4; ++t) {
            // B fragment: lane holds column n0+t*16+l15; K run contiguous in WT
            v16bf bf = *reinterpret_cast<const v16bf*>(
                BT + (size_t)(n0 + t * 16 + l15) * EMBED + kc + 16 * hi);
            acc[0][t] = __builtin_amdgcn_wmma_f32_16x16x32_bf16(
                false, af0, false, bf, (short)0, acc[0][t], false, false);
            acc[1][t] = __builtin_amdgcn_wmma_f32_16x16x32_bf16(
                false, af1, false, bf, (short)0, acc[1][t], false, false);
        }
    }

#pragma unroll
    for (int u = 0; u < 2; ++u)
#pragma unroll
        for (int t = 0; t < 4; ++t) {
            const int col = n0 + t * 16 + l15;
            const float b = bias[col];
#pragma unroll
            for (int r = 0; r < 8; ++r) {
                const int row = m0 + u * 16 + r + 8 * hi;
                const float v = acc[u][t][r] + b;
                if (mode == 0) {
                    ((unsigned short*)Out)[(size_t)row * EMBED + col] = f2bf(v);
                } else if (mode == 1) {
                    ((unsigned short*)Out)[(size_t)col * SEQL + row] = f2bf(v);
                } else {
                    ((float*)Out)[(size_t)row * EMBED + col] = v;
                }
            }
        }
}

// ---------------------------------------------------------------------------
// Windowed attention: one wave per (16 query rows, head).
// scores = Q*K^T/sqrt(n) over 144-col window union, mask, softmax (f32),
// LDS transpose of P to A-layout, ctx = P*V via WMMA, scale by 1/rowsum.
// ---------------------------------------------------------------------------
__global__ __launch_bounds__(32)
void attn_wmma(const unsigned short* __restrict__ Q,   // [SEQL][EMBED] bf16
               const unsigned short* __restrict__ K,   // [SEQL][EMBED] bf16
               const unsigned short* __restrict__ Vt,  // [EMBED][SEQL] bf16
               unsigned short* __restrict__ Ctx) {     // [SEQL][EMBED] bf16
    __shared__ float lds[16 * LDSTR];

    const int lane = threadIdx.x & 31;
    const int l15  = lane & 15;
    const int hi   = (lane >> 4) & 1;
    const int m0   = blockIdx.x * 16;
    const int h    = blockIdx.y;
    const float scale = 0.022097086912079612f;  // 1/sqrt(2048)

    // Q A-fragments: 16x64 tile = 2 K-chunks of 32
    v16bf aq[2];
    const int qrow = (m0 + l15) * EMBED + h * HD;
#pragma unroll
    for (int c = 0; c < 2; ++c) aq[c] = load_afrag(Q, qrow, c * 32, hi);

    const int cstart = m0 - SPAN;

    // ---- scores: 9 tiles of 16 cols, each K=64 -> 2 WMMAs ----
    v8f s[NWTILE];
#pragma unroll
    for (int j = 0; j < NWTILE; ++j) {
        const int p  = cstart + j * 16 + l15;          // this lane's key pos
        const int pc = min(max(p, 0), SEQL - 1);       // clamped (masked later)
        const unsigned short* kb = K + (size_t)pc * EMBED + h * HD;
        v8f a = zero8();
#pragma unroll
        for (int c = 0; c < 2; ++c) {
            v16bf bf = *reinterpret_cast<const v16bf*>(kb + c * 32 + 16 * hi);
            a = __builtin_amdgcn_wmma_f32_16x16x32_bf16(
                false, aq[c], false, bf, (short)0, a, false, false);
        }
        s[j] = a;
    }

    // ---- mask + scale + row max (rows r+8*hi live across 16-lane halves) ----
    float rmax[8];
#pragma unroll
    for (int r = 0; r < 8; ++r) rmax[r] = -__builtin_inff();
#pragma unroll
    for (int j = 0; j < NWTILE; ++j) {
        const int p = cstart + j * 16 + l15;
#pragma unroll
        for (int r = 0; r < 8; ++r) {
            const int m = m0 + r + 8 * hi;
            const bool valid = (p >= 0) && (p < SEQL) &&
                               (p >= m - SPAN) && (p <= m + SPAN);
            const float v = valid ? s[j][r] * scale : -__builtin_inff();
            s[j][r] = v;
            rmax[r] = fmaxf(rmax[r], v);
        }
    }
#pragma unroll
    for (int r = 0; r < 8; ++r) {
        rmax[r] = fmaxf(rmax[r], __shfl_xor(rmax[r], 1));
        rmax[r] = fmaxf(rmax[r], __shfl_xor(rmax[r], 2));
        rmax[r] = fmaxf(rmax[r], __shfl_xor(rmax[r], 4));
        rmax[r] = fmaxf(rmax[r], __shfl_xor(rmax[r], 8));
    }

    // ---- exp + row sum ----
    float rsum[8];
#pragma unroll
    for (int r = 0; r < 8; ++r) rsum[r] = 0.0f;
#pragma unroll
    for (int j = 0; j < NWTILE; ++j) {
#pragma unroll
        for (int r = 0; r < 8; ++r) {
            const float e = __expf(s[j][r] - rmax[r]);
            s[j][r] = e;
            rsum[r] += e;
        }
    }
    float rinv[8];
#pragma unroll
    for (int r = 0; r < 8; ++r) {
        rsum[r] += __shfl_xor(rsum[r], 1);
        rsum[r] += __shfl_xor(rsum[r], 2);
        rsum[r] += __shfl_xor(rsum[r], 4);
        rsum[r] += __shfl_xor(rsum[r], 8);
        rinv[r] = 1.0f / rsum[r];
    }

    // ---- LDS transpose: C-layout (lane=col) -> A-layout (lane=row) ----
#pragma unroll
    for (int j = 0; j < NWTILE; ++j)
#pragma unroll
        for (int r = 0; r < 8; ++r)
            lds[(r + 8 * hi) * LDSTR + j * 16 + l15] = s[j][r];
#pragma unroll
    for (int r = 0; r < 8; ++r)                 // zero K-padding 144..159
        lds[(r + 8 * hi) * LDSTR + 144 + l15] = 0.0f;
    __syncthreads();   // single-wave WG: S_NOP, but orders the DS ops

    // ---- ctx = P * Vwin : 5 K-chunks x 4 N-tiles of 16 dims ----
    v8f acc[4];
#pragma unroll
    for (int t = 0; t < 4; ++t) acc[t] = zero8();

#pragma unroll
    for (int c = 0; c < 5; ++c) {
        BF16x16 ua;
        const int wlo = c * 32 + 16 * hi;
#pragma unroll
        for (int i = 0; i < 8; ++i) {
            ua.s[i]     = f2bf(lds[l15 * LDSTR + wlo + i]);
            ua.s[8 + i] = f2bf(lds[l15 * LDSTR + wlo + 8 + i]);
        }
        const int base = cstart + c * 32 + 16 * hi;   // multiple of 16
#pragma unroll
        for (int t = 0; t < 4; ++t) {
            const int d = t * 16 + l15;
            const unsigned short* vrow = Vt + (size_t)(h * HD + d) * SEQL;
            v16bf bf;
            if (base >= 0 && base + 16 <= SEQL) {
                bf = *reinterpret_cast<const v16bf*>(vrow + base);
            } else {
                BF16x16 ub;                 // edge tiles: OOB lanes hit P==0
#pragma unroll
                for (int e = 0; e < 16; ++e) {
                    int p = min(max(base + e, 0), SEQL - 1);
                    ub.s[e] = vrow[p];
                }
                bf = ub.v;
            }
            acc[t] = __builtin_amdgcn_wmma_f32_16x16x32_bf16(
                false, ua.v, false, bf, (short)0, acc[t], false, false);
        }
    }

    // ---- normalize and store ctx (bf16 row-major) ----
#pragma unroll
    for (int t = 0; t < 4; ++t)
#pragma unroll
        for (int r = 0; r < 8; ++r) {
            const float v = acc[t][r] * rinv[r];
            Ctx[(size_t)(m0 + r + 8 * hi) * EMBED + h * HD + t * 16 + l15] = f2bf(v);
        }
}

// ---------------------------------------------------------------------------
extern "C" void kernel_launch(void* const* d_in, const int* in_sizes, int n_in,
                              void* d_out, int out_size, void* d_ws, size_t ws_size,
                              hipStream_t stream) {
    const float* query = (const float*)d_in[0];
    const float* key   = (const float*)d_in[1];
    const float* value = (const float*)d_in[2];
    const float* Wq    = (const float*)d_in[3];
    const float* bq    = (const float*)d_in[4];
    const float* Wk    = (const float*)d_in[5];
    const float* bk    = (const float*)d_in[6];
    const float* Wv    = (const float*)d_in[7];
    const float* bv    = (const float*)d_in[8];
    const float* Wo    = (const float*)d_in[9];
    const float* bo    = (const float*)d_in[10];

    uint8_t* ws = (uint8_t*)d_ws;
    size_t off = 0;
    auto carve = [&](size_t bytes) {
        void* p = ws + off;
        off += (bytes + 255) & ~(size_t)255;
        return p;
    };
    const size_t ACT = (size_t)SEQL * EMBED;   // activation elems
    const size_t WEL = (size_t)EMBED * EMBED;  // weight elems

    unsigned short* qb  = (unsigned short*)carve(ACT * 2);
    unsigned short* kb  = (unsigned short*)carve(ACT * 2);
    unsigned short* vb  = (unsigned short*)carve(ACT * 2);
    unsigned short* WqT = (unsigned short*)carve(WEL * 2);
    unsigned short* WkT = (unsigned short*)carve(WEL * 2);
    unsigned short* WvT = (unsigned short*)carve(WEL * 2);
    unsigned short* WoT = (unsigned short*)carve(WEL * 2);
    unsigned short* Qb  = (unsigned short*)carve(ACT * 2);
    unsigned short* Kb  = (unsigned short*)carve(ACT * 2);
    unsigned short* Vt  = (unsigned short*)carve(ACT * 2);   // [EMBED][SEQL]
    unsigned short* Ctx = (unsigned short*)carve(ACT * 2);

    // 1) precision convert + weight transpose
    const int na = (int)ACT, nw = (int)WEL;
    cvt_f32_bf16<<<(na + 255) / 256, 256, 0, stream>>>(query, qb, na);
    cvt_f32_bf16<<<(na + 255) / 256, 256, 0, stream>>>(key,   kb, na);
    cvt_f32_bf16<<<(na + 255) / 256, 256, 0, stream>>>(value, vb, na);
    transpose_f32_bf16<<<(nw + 255) / 256, 256, 0, stream>>>(Wq, WqT, EMBED, EMBED);
    transpose_f32_bf16<<<(nw + 255) / 256, 256, 0, stream>>>(Wk, WkT, EMBED, EMBED);
    transpose_f32_bf16<<<(nw + 255) / 256, 256, 0, stream>>>(Wv, WvT, EMBED, EMBED);
    transpose_f32_bf16<<<(nw + 255) / 256, 256, 0, stream>>>(Wo, WoT, EMBED, EMBED);

    // 2) QKV projections (WMMA): grid = (M/32/4, N/64)
    dim3 gg(SEQL / 32 / 4, EMBED / 64);     // (16, 12)
    gemm_wmma_bf16<<<gg, 128, 0, stream>>>(qb, WqT, bq, Qb, 0);
    gemm_wmma_bf16<<<gg, 128, 0, stream>>>(kb, WkT, bk, Kb, 0);
    gemm_wmma_bf16<<<gg, 128, 0, stream>>>(vb, WvT, bv, Vt, 1);

    // 3) windowed attention (WMMA + f32 softmax)
    attn_wmma<<<dim3(SEQL / 16, HEADS), 32, 0, stream>>>(Qb, Kb, Vt, Ctx);

    // 4) output projection, f32 result
    gemm_wmma_bf16<<<gg, 128, 0, stream>>>(Ctx, WoT, bo, d_out, 2);
}
